// DynamicGC_4698694221988
// MI455X (gfx1250) — compile-verified, matrix operand order checked by reading, not run on hardware
//
#include <hip/hip_runtime.h>

// ---------------------------------------------------------------------------
// Sizes from the reference: B=1024, N=400, F=64, UNITS=64
// NP = 416 : N padded up to a multiple of 32 (WMMA bf16 K-step) for K loops.
// ---------------------------------------------------------------------------
#define BN    1024
#define NN    400
#define NP    416
#define FF    64
#define UU    64
#define NEGV  (-1e16f)

typedef __bf16 v16bf __attribute__((ext_vector_type(16)));
typedef __bf16 v8bf  __attribute__((ext_vector_type(8)));
typedef float  v8f   __attribute__((ext_vector_type(8)));
typedef unsigned int u32x4 __attribute__((ext_vector_type(4)));
typedef int          i32x4 __attribute__((ext_vector_type(4)));
typedef int          i32x8 __attribute__((ext_vector_type(8)));

// LDS layout byte offsets (dynamic shared, no static __shared__ -> base 0)
#define LDS_XT     0u        // bf16 [64][416]
#define LDS_AWT    53248u    // bf16 [16][416]  (stage-1 A, reused as attn)
#define LDS_W2T    66560u    // bf16 [400][64]  (stage-2 B, reused as w3t)
#define LDS_DENSE  117760u   // f32  [16][416]
#define LDS_FEAT   144384u   // bf16 [16][64]
#define LDS_NODE   146432u   // bf16 [16][64]
#define LDS_TOTAL  148480u

// ---------------------------------------------------------------------------
// CDNA5 async global->LDS 128-bit copy (§15.18.3 opcode 98, ASYNCcnt-tracked).
// VDST VGPR carries the per-lane LDS byte address; VADDR the 64-bit global.
// ---------------------------------------------------------------------------
__device__ __forceinline__ void async_copy_b128(unsigned lds_byte_off, const void* gsrc) {
    asm volatile("global_load_async_to_lds_b128 %0, %1, off"
                 :: "v"(lds_byte_off), "v"(gsrc)
                 : "memory");
}

__device__ __forceinline__ void wait_async0() {
#if __has_builtin(__builtin_amdgcn_s_wait_asynccnt)
    __builtin_amdgcn_s_wait_asynccnt(0);
#else
    asm volatile("s_wait_asynccnt 0x0" ::: "memory");
#endif
}

// ---------------------------------------------------------------------------
// bf16 16x32 WMMA operand fragment loader (A or B operand; same striping).
// Layout per CDNA5 ISA 7.12.2:
//   lanes 0-15 : row/col = lane, elems 0..7 -> K=kt*32+0..7,  8..15 -> +16..23
//   lanes16-31 : row/col = lane-16, elems 0..7 -> +8..15,     8..15 -> +24..31
// `base` points at element [0][0] of a row-major [16][ldk] bf16 tile in LDS.
// Both 8-element chunks are contiguous -> two ds_load_b128 per fragment.
// ---------------------------------------------------------------------------
__device__ __forceinline__ v16bf load_frag(const __bf16* base, int ldk, int kt) {
    const int lane = threadIdx.x & 31;
    const int half = lane >> 4;
    const int idx  = lane & 15;
    const __bf16* p = base + idx * ldk + kt * 32 + half * 8;
    v8bf lo = *(const v8bf*)(p);        // K = kt*32 + half*8 .. +7
    v8bf hi = *(const v8bf*)(p + 16);   // K = kt*32 + 16 + half*8 .. +7
    v16bf r;
#pragma unroll
    for (int i = 0; i < 8; ++i) { r[i] = lo[i]; r[i + 8] = hi[i]; }
    return r;
}

// ---------------------------------------------------------------------------
// Prep kernel: convert weights to bf16 in workspace, in WMMA-friendly order.
//   ws[0              .. 400*416)      : awt[m][n] = A[n][m]*W1[n][m] (n>=400 -> 0)
//   ws[400*416        .. +400*64)      : w2t[j][f] = W2[f][j]
//   ws[400*416+400*64 .. +64*64)       : w3t[u][f] = W3[f][u]
// ---------------------------------------------------------------------------
__global__ void prep_weights(const float* __restrict__ A,
                             const float* __restrict__ W1,
                             const float* __restrict__ W2,
                             const float* __restrict__ W3,
                             __bf16* __restrict__ ws) {
    int t = blockIdx.x * 256 + threadIdx.x;
    const int AWT = NN * NP;
    const int W2T = NN * FF;
    const int W3T = UU * FF;
    if (t < AWT) {
        int m = t / NP, n = t % NP;
        float v = 0.0f;
        if (n < NN) v = A[n * NN + m] * W1[n * NN + m];
        ws[t] = (__bf16)v;
    } else if (t < AWT + W2T) {
        int r = t - AWT;
        int j = r / FF, f = r % FF;
        ws[t] = (__bf16)W2[f * NN + j];
    } else if (t < AWT + W2T + W3T) {
        int r = t - AWT - W2T;
        int u = r / FF, f = r % FF;
        ws[t] = (__bf16)W3[f * UU + u];
    }
}

// ---------------------------------------------------------------------------
// Fused kernel. One workgroup = (batch b, 16-row tile m0..m0+15).
// 128 threads = 4 wave32s. Four chained WMMA GEMM stages + masked softmax.
// Weight staging uses TDM / async-to-LDS so it overlaps the f32->bf16
// x-transpose fill (stage 0) and the stage-4 WMMA loop (w3t overlay).
// ---------------------------------------------------------------------------
__global__ __launch_bounds__(128) void gat_fused(const float* __restrict__ x,
                                                 const float* __restrict__ A,
                                                 const float* __restrict__ b1,
                                                 const float* __restrict__ b2,
                                                 const __bf16* __restrict__ ws,
                                                 float* __restrict__ out) {
    extern __shared__ char smem[];
    __bf16* xT    = (__bf16*)(smem + LDS_XT);
    __bf16* awt   = (__bf16*)(smem + LDS_AWT);   // also attn
    __bf16* w2t   = (__bf16*)(smem + LDS_W2T);   // also w3t
    float*  dense = (float*) (smem + LDS_DENSE);
    __bf16* feat  = (__bf16*)(smem + LDS_FEAT);
    __bf16* node  = (__bf16*)(smem + LDS_NODE);

    const int tid  = threadIdx.x;
    const int wave = tid >> 5;
    const int lane = tid & 31;
    const int half = lane >> 4;
    const int hrow = half * 8;
    const int idx  = lane & 15;

    const int b  = blockIdx.x / (NN / 16);
    const int mt = blockIdx.x % (NN / 16);
    const int m0 = mt * 16;

    const float* xb = x + (size_t)b * NN * FF;

    // ---- Stage 0a: kick off weight staging (async/TDM) ---------------------
    // awt rows m0..m0+15: 16x416 bf16 = 832 x 16B chunks, async to LDS.
    {
        const char* asrc = (const char*)(ws + m0 * NP);
        for (int i = tid; i < 832; i += 128)
            async_copy_b128(LDS_AWT + (unsigned)i * 16u, asrc + (size_t)i * 16);
    }
#if __has_builtin(__builtin_amdgcn_tensor_load_to_lds)
    // w2t: one TDM descriptor moves the whole 51200B block (6400 x 8B elems).
    if (wave == 0) {
        unsigned long long ga = (unsigned long long)(uintptr_t)(ws + NN * NP);
        u32x4 g0 = { 1u,                                   // count=1 (valid D#)
                     LDS_W2T,                              // lds_addr
                     (unsigned)(ga & 0xffffffffu),         // global_addr[31:0]
                     (unsigned)(((ga >> 32) & 0x1ffffffu)  // global_addr[56:32]
                                | 0x80000000u) };          // type=2 ("image")
        i32x8 g1 = { 0x00030000,     // data_size=3 (8B), wg_mask=0
                     0x19000000,     // tensor_dim0[15:0]=6400 in bits 63:48
                     0x00010000,     // tensor_dim0[31:16]=0, tensor_dim1=1
                     0x19000000,     // tile_dim0=6400 in bits 127:112
                     0,              // tile_dim1=0, tile_dim2=0 (unused)
                     6400,           // tensor_dim0_stride[31:0]
                     0, 0 };
        i32x4 gz4 = { 0, 0, 0, 0 };
        i32x8 gz8 = { 0, 0, 0, 0, 0, 0, 0, 0 };
        __builtin_amdgcn_tensor_load_to_lds(g0, g1, gz4, gz4, gz8, 0);
    }
#else
    {   // fallback: async-b128 copy of w2t (3200 x 16B chunks)
        const char* wsrc = (const char*)(ws + NN * NP);
        for (int i = tid; i < 3200; i += 128)
            async_copy_b128(LDS_W2T + (unsigned)i * 16u, wsrc + (size_t)i * 16);
    }
#endif

    // ---- Stage 0b: xT fill (overlaps the async/TDM weight staging) ---------
    // Coalesced float4 reads of x[b][n][f], transposed bf16 scatter to LDS.
    for (int i = tid; i < NN * (FF / 4); i += 128) {
        int n  = i >> 4;
        int f4 = (i & 15) * 4;
        float4 v = *(const float4*)(xb + n * FF + f4);
        xT[(f4 + 0) * NP + n] = (__bf16)v.x;
        xT[(f4 + 1) * NP + n] = (__bf16)v.y;
        xT[(f4 + 2) * NP + n] = (__bf16)v.z;
        xT[(f4 + 3) * NP + n] = (__bf16)v.w;
    }
    for (int i = tid; i < FF * (NP - NN); i += 128) {          // zero K pad
        int f = i / (NP - NN);
        int n = NN + i % (NP - NN);
        xT[f * NP + n] = (__bf16)0.0f;
    }

    wait_async0();
#if __has_builtin(__builtin_amdgcn_tensor_load_to_lds)
    if (wave == 0) __builtin_amdgcn_s_wait_tensorcnt(0);
#endif
    __syncthreads();

    // ---- Stage 1: feature = awt(16x416) * xT^T  -> feat (16x64) -----------
    {
        v8f acc = {};
        const __bf16* bb = xT + wave * 16 * NP;   // B cols f = wave*16 + idx
        for (int kt = 0; kt < NP / 32; ++kt) {
            v16bf a  = load_frag(awt, NP, kt);
            v16bf bf = load_frag(bb, NP, kt);
            acc = __builtin_amdgcn_wmma_f32_16x16x32_bf16(
                false, a, false, bf, (short)0, acc, false, false);
        }
#pragma unroll
        for (int r = 0; r < 8; ++r)
            feat[(hrow + r) * FF + wave * 16 + idx] = (__bf16)acc[r];
    }
    __syncthreads();

    // ---- Stage 2: dense = feat(16x64) * W2 + b1 + mask  -> dense (16x400) --
    for (int jt = wave; jt < NN / 16; jt += 4) {
        v8f acc = {};
        for (int kt = 0; kt < FF / 32; ++kt) {
            v16bf a  = load_frag(feat, FF, kt);
            v16bf bf = load_frag(w2t + jt * 16 * FF, FF, kt);
            acc = __builtin_amdgcn_wmma_f32_16x16x32_bf16(
                false, a, false, bf, (short)0, acc, false, false);
        }
        const int j  = jt * 16 + idx;
        const float bj = b1[j];
#pragma unroll
        for (int r = 0; r < 8; ++r) {
            const int m = m0 + hrow + r;
            const float mask = A[m * NN + j];          // L2-resident
            dense[(hrow + r) * NP + j] = acc[r] + bj + NEGV * (1.0f - mask);
        }
    }
    __syncthreads();

    // ---- Stage 3: masked softmax over rows of dense; attn (bf16) -> awt ----
    {
        const int row = tid >> 3;      // 16 rows x 8 lanes; 8-groups contiguous
        const int sub = tid & 7;       // within one wave -> shfl_xor works
        float* drow = dense + row * NP;
        float mx = -3.0e38f;
        for (int j = sub; j < NN; j += 8) mx = fmaxf(mx, drow[j]);
        mx = fmaxf(mx, __shfl_xor(mx, 1));
        mx = fmaxf(mx, __shfl_xor(mx, 2));
        mx = fmaxf(mx, __shfl_xor(mx, 4));
        float s = 0.0f;
        for (int j = sub; j < NN; j += 8) {
            float e = __expf(drow[j] - mx);
            drow[j] = e;
            s += e;
        }
        s += __shfl_xor(s, 1);
        s += __shfl_xor(s, 2);
        s += __shfl_xor(s, 4);
        const float inv = 1.0f / s;
        __bf16* arow = awt + row * NP;                 // reuse awt as attn
        for (int j = sub; j < NN; j += 8) arow[j] = (__bf16)(drow[j] * inv);
        if (sub == 0)
            for (int j = NN; j < NP; ++j) arow[j] = (__bf16)0.0f;
    }
    __syncthreads();

    // ---- w3t overlay: async copy issued now, waited after stage 4 ----------
    {
        const char* w3src = (const char*)(ws + NN * NP + NN * FF);
        for (int i = tid; i < 512; i += 128)
            async_copy_b128(LDS_W2T + (unsigned)i * 16u, w3src + (size_t)i * 16);
    }

    // ---- Stage 4: node = attn(16x416) * xT^T -> node (16x64) ---------------
    {
        v8f acc = {};
        const __bf16* bb = xT + wave * 16 * NP;
        for (int kt = 0; kt < NP / 32; ++kt) {
            v16bf a  = load_frag(awt, NP, kt);
            v16bf bf = load_frag(bb, NP, kt);
            acc = __builtin_amdgcn_wmma_f32_16x16x32_bf16(
                false, a, false, bf, (short)0, acc, false, false);
        }
#pragma unroll
        for (int r = 0; r < 8; ++r)
            node[(hrow + r) * FF + wave * 16 + idx] = (__bf16)acc[r];
    }
    wait_async0();          // w3t overlay complete (hidden under stage 4)
    __syncthreads();

    // ---- Stage 5: out = node(16x64) * W3 + b2 ------------------------------
    {
        v8f acc = {};
        for (int kt = 0; kt < FF / 32; ++kt) {
            v16bf a  = load_frag(node, FF, kt);
            v16bf bf = load_frag(w2t + wave * 16 * FF, FF, kt);  // w3t overlay
            acc = __builtin_amdgcn_wmma_f32_16x16x32_bf16(
                false, a, false, bf, (short)0, acc, false, false);
        }
        const int u  = wave * 16 + idx;
        const float bu = b2[u];
        float* ob = out + (size_t)b * NN * UU + (size_t)m0 * UU;
#pragma unroll
        for (int r = 0; r < 8; ++r)
            ob[(hrow + r) * UU + u] = acc[r] + bu;
    }
}

// ---------------------------------------------------------------------------
extern "C" void kernel_launch(void* const* d_in, const int* in_sizes, int n_in,
                              void* d_out, int out_size, void* d_ws, size_t ws_size,
                              hipStream_t stream) {
    const float* x  = (const float*)d_in[0];
    const float* A  = (const float*)d_in[1];
    const float* W1 = (const float*)d_in[2];
    const float* W2 = (const float*)d_in[3];
    const float* W3 = (const float*)d_in[4];
    const float* b1 = (const float*)d_in[5];
    const float* b2 = (const float*)d_in[6];
    float* out = (float*)d_out;
    __bf16* ws = (__bf16*)d_ws;   // needs 392192 bytes of bf16 weight staging

    const int prep_elems = NN * NP + NN * FF + UU * FF;   // 196096
    prep_weights<<<(prep_elems + 255) / 256, 256, 0, stream>>>(A, W1, W2, W3, ws);

    const size_t shmem = LDS_TOTAL;                        // dynamic LDS bytes
    gat_fused<<<BN * (NN / 16), 128, shmem, stream>>>(x, A, b1, b2, ws, out);
}
